// KeyValueMemory_39204461478026
// MI455X (gfx1250) — matrix operand order repaired and checked by reading, not compiled, not run on hardware
//
#include <hip/hip_runtime.h>
#include <hip/hip_bf16.h>

typedef _Float16 f16;
typedef __attribute__((ext_vector_type(16))) _Float16 v16h;
typedef __attribute__((ext_vector_type(8)))  float    v8f;

#define B_   256
#define C_   512
#define HW_  784
#define M_   65536
#define K_   32
#define NEG_INF (-3.402823466e38f)

#define QS_LD 40            // padded row stride (halves): 80 B, 16B-aligned
#define KS_LD 72            // padded row stride (halves): 144 B, 16B-aligned
#define QS_SZ (256 * QS_LD) // one query buffer (halves)
#define KS_SZ (32 * KS_LD)  // one key buffer (halves)

// ---------------- 1) Global average pool: x[B,C,784] -> q[B*C] ----------------
__global__ __launch_bounds__(256) void gap_kernel(const float* __restrict__ x,
                                                  float* __restrict__ q) {
    __shared__ float red[256];
    const int bc = blockIdx.x;                 // 0 .. B*C-1
    const float* p = x + (size_t)bc * HW_;
    float s = 0.f;
    for (int i = threadIdx.x; i < HW_; i += 256) s += p[i];
    red[threadIdx.x] = s;
    __syncthreads();
    for (int o = 128; o > 0; o >>= 1) {
        if (threadIdx.x < o) red[threadIdx.x] += red[threadIdx.x + o];
        __syncthreads();
    }
    if (threadIdx.x == 0) q[bc] = red[0] * (1.0f / (float)HW_);
}

// ---------------- 2) Normalize query rows, emit f16 [B,512] ----------------
__global__ __launch_bounds__(256) void qnorm_kernel(const float* __restrict__ q,
                                                    f16* __restrict__ qn16) {
    __shared__ float red[256];
    __shared__ float inv;
    const int b = blockIdx.x;
    const float* row = q + (size_t)b * C_;
    float v0 = row[threadIdx.x * 2 + 0];
    float v1 = row[threadIdx.x * 2 + 1];
    red[threadIdx.x] = v0 * v0 + v1 * v1;
    __syncthreads();
    for (int o = 128; o > 0; o >>= 1) {
        if (threadIdx.x < o) red[threadIdx.x] += red[threadIdx.x + o];
        __syncthreads();
    }
    if (threadIdx.x == 0) inv = 1.0f / fmaxf(sqrtf(red[0]), 1e-12f);
    __syncthreads();
    qn16[(size_t)b * C_ + threadIdx.x * 2 + 0] = (f16)(v0 * inv);
    qn16[(size_t)b * C_ + threadIdx.x * 2 + 1] = (f16)(v1 * inv);
}

// ---------------- 3) Per-key inverse norms ----------------
__global__ __launch_bounds__(256) void knorm_kernel(const float* __restrict__ keys,
                                                    float* __restrict__ knorm) {
    __shared__ float red[256];
    const int m = blockIdx.x;
    const float* row = keys + (size_t)m * C_;
    float v0 = row[threadIdx.x * 2 + 0];
    float v1 = row[threadIdx.x * 2 + 1];
    red[threadIdx.x] = v0 * v0 + v1 * v1;
    __syncthreads();
    for (int o = 128; o > 0; o >>= 1) {
        if (threadIdx.x < o) red[threadIdx.x] += red[threadIdx.x + o];
        __syncthreads();
    }
    if (threadIdx.x == 0) knorm[m] = 1.0f / fmaxf(sqrtf(red[0]), 1e-12f);
}

// ---------------- 4) sim = q_norm @ k_norm^T via WMMA f16 ----------------
union V16 { v16h h; uint4 q[2]; };

// A fragment: 16x32 f16. Lane L: row M = L&15, kofs = 8*(L>=16).
__device__ __forceinline__ v16h load_a_frag(const f16* __restrict__ qs, int row, int lane) {
    const f16* p = qs + row * QS_LD;
    const int kofs = (lane >> 4) << 3;   // 0 or 8
    V16 a;
    a.q[0] = *(const uint4*)(p + kofs);
    a.q[1] = *(const uint4*)(p + 16 + kofs);
    return a.h;
}

// B fragment: 32x16 f16 (KxN). Lane L holds N = n0+(L&15); b[i] = B[kofs+i][N], b[8+i] = B[16+kofs+i][N].
__device__ __forceinline__ v16h load_b_frag(const f16* __restrict__ kst, int n0, int lane) {
    const f16* p = kst + n0 + (lane & 15);
    const int kofs = (lane >> 4) << 3;   // 0 or 8
    V16 b;
#pragma unroll
    for (int i = 0; i < 8; ++i) {
        b.h[i]     = p[(kofs + i) * KS_LD];
        b.h[8 + i] = p[(16 + kofs + i) * KS_LD];
    }
    return b.h;
}

__global__ __launch_bounds__(256, 2) void sim_gemm_kernel(const f16* __restrict__ qn16,
                                                          const float* __restrict__ keys,
                                                          const float* __restrict__ knorm,
                                                          float* __restrict__ sim) {
    __shared__ f16 qs[2 * QS_SZ];    // 40 KB: double-buffered query chunks
    __shared__ f16 kst[2 * KS_SZ];   // 9 KB: double-buffered transposed key chunks

    const int tid   = threadIdx.x;
    const int lane  = tid & 31;
    const int wid   = tid >> 5;          // 8 waves
    const int m0    = blockIdx.x * 64;   // key tile base
    const int bbase = wid * 32;          // each wave: 32 query rows

    v8f acc[2][4];
#pragma unroll
    for (int sb = 0; sb < 2; ++sb)
#pragma unroll
        for (int sm = 0; sm < 4; ++sm)
#pragma unroll
            for (int r = 0; r < 8; ++r) acc[sb][sm][r] = 0.0f;

    const int kr   = tid >> 2;           // key row within tile: 0..63
    const int kcol = (tid & 3) * 8;      // K sub-offset: 0,8,16,24
    const float kscale = knorm[m0 + kr];
    const float* krow = keys + (size_t)(m0 + kr) * C_;
    const char*  qsrc0 = (const char*)(qn16 + (size_t)tid * C_);  // this thread's query row

    // register pipeline for the HBM key stream only
    float kreg[8];
#pragma unroll
    for (int j = 0; j < 8; ++j) kreg[j] = krow[kcol + j];

#pragma unroll 1
    for (int kc = 0; kc < C_ / 32; ++kc) {   // 16 K-chunks
        f16* qsb  = qs  + (kc & 1) * QS_SZ;
        f16* kstb = kst + (kc & 1) * KS_SZ;

        // ---- stage chunk kc into LDS buffer (kc&1) ----
        // queries: async global->LDS copy (64 B per thread, no VGPR round-trip)
        {
            unsigned    ldst = (unsigned)(size_t)(void*)(qsb + tid * QS_LD);
            const char* gsrc = qsrc0 + (size_t)kc * 64;
            asm volatile(
                "global_load_async_to_lds_b128 %0, %1, off\n\t"
                "global_load_async_to_lds_b128 %0, %1, off offset:16\n\t"
                "global_load_async_to_lds_b128 %0, %1, off offset:32\n\t"
                "global_load_async_to_lds_b128 %0, %1, off offset:48"
                :: "v"(ldst), "v"(gsrc) : "memory");
        }
        // keys: from pipelined registers, normalize + cvt, transposed scatter
#pragma unroll
        for (int j = 0; j < 8; ++j)
            kstb[(kcol + j) * KS_LD + kr] = (f16)(kreg[j] * kscale);

        // preload next key chunk (overlaps with waits + compute below)
        if (kc < 15) {
            const int nc = kc + 1;
#pragma unroll
            for (int j = 0; j < 8; ++j) kreg[j] = krow[nc * 32 + kcol + j];
            if (kc < 14)
                __builtin_prefetch(krow + (kc + 2) * 32 + kcol, 0, 1);  // global_prefetch_b8
        }

        asm volatile("s_wait_asynccnt 0x0" ::: "memory");  // my LDS writes landed
        __syncthreads();   // single barrier per iteration (double-buffered)

        // ---- compute on buffer (kc&1) ----
        v16h bf[4];
#pragma unroll
        for (int sm = 0; sm < 4; ++sm) bf[sm] = load_b_frag(kstb, sm * 16, lane);
#pragma unroll
        for (int sb = 0; sb < 2; ++sb) {
            v16h a = load_a_frag(qsb, bbase + sb * 16 + (lane & 15), lane);
#pragma unroll
            for (int sm = 0; sm < 4; ++sm) {
                acc[sb][sm] = __builtin_amdgcn_wmma_f32_16x16x32_f16(
                    false, a, false, bf[sm], (short)0, acc[sb][sm], false, false);
            }
        }
    }

    // store D tiles: VGPR r, lane L -> row = r + 8*(L>=16), col = L&15
#pragma unroll
    for (int sb = 0; sb < 2; ++sb)
#pragma unroll
        for (int sm = 0; sm < 4; ++sm)
#pragma unroll
            for (int r = 0; r < 8; ++r) {
                const int row = bbase + sb * 16 + r + ((lane >> 4) << 3);
                const int col = m0 + sm * 16 + (lane & 15);
                sim[(size_t)row * M_ + col] = acc[sb][sm][r];
            }
}

// ---------------- 5) top-32 + softmax + weighted value gather ----------------
__global__ __launch_bounds__(256) void topk_softmax_gather_kernel(
    const float* __restrict__ sim, const float* __restrict__ values,
    float* __restrict__ matched) {
    const int b   = blockIdx.x;
    const int tid = threadIdx.x;

    __shared__ float tval[256];
    __shared__ int   tm[256];
    __shared__ int   tpos[256];
    __shared__ float topv[K_];
    __shared__ int   topm[K_];
    __shared__ float wts[K_];
    __shared__ int   winner;

    // per-thread top-32 over strided slice
    float lv[K_]; int lm[K_];
#pragma unroll
    for (int i = 0; i < K_; ++i) { lv[i] = NEG_INF; lm[i] = 0; }
    float curmin = NEG_INF; int curminpos = 0;

    const float* srow = sim + (size_t)b * M_;
    for (int m = tid; m < M_; m += 256) {
        const float v = srow[m];
        if (v > curmin) {
            lv[curminpos] = v; lm[curminpos] = m;
            curmin = lv[0]; curminpos = 0;
#pragma unroll
            for (int i = 1; i < K_; ++i)
                if (lv[i] < curmin) { curmin = lv[i]; curminpos = i; }
        }
    }
    // thread-local best
    {
        float bv = lv[0]; int bp = 0;
#pragma unroll
        for (int i = 1; i < K_; ++i) if (lv[i] > bv) { bv = lv[i]; bp = i; }
        tval[tid] = bv; tm[tid] = lm[bp]; tpos[tid] = bp;
    }
    __syncthreads();

    // 32 rounds of global extraction
    for (int k = 0; k < K_; ++k) {
        if (tid == 0) {
            float best = tval[0]; int w = 0;
            for (int t = 1; t < 256; ++t)
                if (tval[t] > best) { best = tval[t]; w = t; }
            topv[k] = best; topm[k] = tm[w]; winner = w;
        }
        __syncthreads();
        if (tid == winner) {
            lv[tpos[tid]] = NEG_INF;
            float nbv = lv[0]; int nbp = 0;
#pragma unroll
            for (int i = 1; i < K_; ++i) if (lv[i] > nbv) { nbv = lv[i]; nbp = i; }
            tval[tid] = nbv; tm[tid] = lm[nbp]; tpos[tid] = nbp;
        }
        __syncthreads();
    }

    // softmax over descending top values (topv[0] is the max)
    if (tid < K_) wts[tid] = expf(topv[tid] - topv[0]);
    __syncthreads();
    if (tid == 0) {
        float s = 0.f;
        for (int i = 0; i < K_; ++i) s += wts[i];
        const float inv = 1.0f / s;
        for (int i = 0; i < K_; ++i) wts[i] *= inv;
    }
    __syncthreads();

    // matched[b,:] = sum_k wts[k] * values[topm[k], :]
    for (int c = tid; c < C_; c += 256) {
        float acc = 0.f;
#pragma unroll 8
        for (int k = 0; k < K_; ++k)
            acc += wts[k] * values[(size_t)topm[k] * C_ + c];
        matched[(size_t)b * C_ + c] = acc;
    }
}

// ---------------- 6) broadcast matched[b,c] over 28x28 ----------------
__global__ __launch_bounds__(256) void broadcast_kernel(const float* __restrict__ matched,
                                                        float* __restrict__ out) {
    const int bc = blockIdx.x;
    const float val = matched[bc];
    float4* o = (float4*)(out + (size_t)bc * HW_);
    if (threadIdx.x < HW_ / 4) {
        float4 v4; v4.x = val; v4.y = val; v4.z = val; v4.w = val;
        o[threadIdx.x] = v4;
    }
}

extern "C" void kernel_launch(void* const* d_in, const int* in_sizes, int n_in,
                              void* d_out, int out_size, void* d_ws, size_t ws_size,
                              hipStream_t stream) {
    const float* x      = (const float*)d_in[0];   // [256,512,28,28]
    const float* keys   = (const float*)d_in[1];   // [65536,512]
    const float* values = (const float*)d_in[2];   // [65536,512]
    // d_in[3] = topk scalar (fixed 32)
    float* out = (float*)d_out;
    char* ws = (char*)d_ws;

    float* q       = (float*)(ws + 0);           // 256*512*4   = 512 KB
    f16*   qn16    = (f16*)  (ws + (512 << 10)); // 256*512*2   = 256 KB
    float* knorm   = (float*)(ws + (768 << 10)); // 65536*4     = 256 KB
    float* matched = (float*)(ws + (1024 << 10));// 256*512*4   = 512 KB
    float* sim     = (float*)(ws + (1536 << 10));// 256*65536*4 =  64 MB

    gap_kernel<<<B_ * C_, 256, 0, stream>>>(x, q);
    qnorm_kernel<<<B_, 256, 0, stream>>>(q, qn16);
    knorm_kernel<<<M_, 256, 0, stream>>>(keys, knorm);
    sim_gemm_kernel<<<M_ / 64, 256, 0, stream>>>(qn16, keys, knorm, sim);
    topk_softmax_gather_kernel<<<B_, 256, 0, stream>>>(sim, values, matched);
    broadcast_kernel<<<B_ * C_, 256, 0, stream>>>(matched, out);
}
